// MemoryAwareContrastiveLoss_10771777978668
// MI455X (gfx1250) — matrix operand order repaired and checked by reading, not compiled, not run on hardware
//
#include <hip/hip_runtime.h>
#include <stdint.h>

// ---------------- problem constants ----------------
#define BB 8192
#define DD 512
#define NCHUNK 4                       // column splits of the 8192-wide sweep
#define COLS_PER_CHUNK (BB / NCHUNK)   // 2048

#define M0F 14.285714285714286f        // 1/TEMPERATURE, >= every logit
#define L2E 1.4426950408889634f
#define LN2F 0.6931471805599453f
#define KSC (M0F * L2E)                // exp(x - M0) = exp2((c-1)*KSC), c = cosine

typedef __attribute__((ext_vector_type(16))) __bf16 v16bf;
typedef __attribute__((ext_vector_type(8)))  float  v8f;

union Frag16 {                         // 32 bytes: one bf16 WMMA A/B fragment per lane
  uint4 q[2];
  v16bf v;
};

// ---------------- kernel 1: normalize rows -> bf16, fp32 pos ----------------
__device__ __forceinline__ unsigned short f2bf_rne(float f) {
  unsigned u = __float_as_uint(f);
  unsigned r = u + 0x7FFFu + ((u >> 16) & 1u);
  return (unsigned short)(r >> 16);
}

__global__ __launch_bounds__(256) void norm_pos_kernel(
    const float* __restrict__ ie, const float* __restrict__ le,
    unsigned short* __restrict__ ieb, unsigned short* __restrict__ leb,
    float* __restrict__ pos) {
  const int lane = threadIdx.x & 31;
  const int wave = threadIdx.x >> 5;
  const int row = blockIdx.x * 8 + wave;   // one wave per row
  const int off = lane * 16;               // 16 contiguous elems per lane

  float xi[16], xl[16];
  const float4* ip = (const float4*)(ie + (size_t)row * DD + off);
  const float4* lp = (const float4*)(le + (size_t)row * DD + off);
#pragma unroll
  for (int q = 0; q < 4; ++q) {
    float4 a = ip[q];
    xi[4 * q + 0] = a.x; xi[4 * q + 1] = a.y; xi[4 * q + 2] = a.z; xi[4 * q + 3] = a.w;
    float4 b = lp[q];
    xl[4 * q + 0] = b.x; xl[4 * q + 1] = b.y; xl[4 * q + 2] = b.z; xl[4 * q + 3] = b.w;
  }
  float ssi = 0.f, ssl = 0.f, dot = 0.f;
#pragma unroll
  for (int e = 0; e < 16; ++e) {
    ssi = fmaf(xi[e], xi[e], ssi);
    ssl = fmaf(xl[e], xl[e], ssl);
    dot = fmaf(xi[e], xl[e], dot);
  }
#pragma unroll
  for (int m = 1; m < 32; m <<= 1) {
    ssi += __shfl_xor(ssi, m, 32);
    ssl += __shfl_xor(ssl, m, 32);
    dot += __shfl_xor(dot, m, 32);
  }
  const float invi = 1.0f / fmaxf(sqrtf(ssi), 1e-12f);
  const float invl = 1.0f / fmaxf(sqrtf(ssl), 1e-12f);
  if (lane == 0) pos[row] = dot * invi * invl * M0F;  // fp32 diagonal logit

  // pack normalized rows as bf16 (RNE)
  unsigned wi[8], wl[8];
#pragma unroll
  for (int p = 0; p < 8; ++p) {
    wi[p] = (unsigned)f2bf_rne(xi[2 * p] * invi) |
            ((unsigned)f2bf_rne(xi[2 * p + 1] * invi) << 16);
    wl[p] = (unsigned)f2bf_rne(xl[2 * p] * invl) |
            ((unsigned)f2bf_rne(xl[2 * p + 1] * invl) << 16);
  }
  uint4* iod = (uint4*)(ieb + (size_t)row * DD + off);
  uint4* lod = (uint4*)(leb + (size_t)row * DD + off);
  iod[0] = make_uint4(wi[0], wi[1], wi[2], wi[3]);
  iod[1] = make_uint4(wi[4], wi[5], wi[6], wi[7]);
  lod[0] = make_uint4(wl[0], wl[1], wl[2], wl[3]);
  lod[1] = make_uint4(wl[4], wl[5], wl[6], wl[7]);
}

// ---------------- kernel 2: fused bf16 WMMA GEMM + masked online sum-exp ----------------
// Each wave: 16 rows, sweeps COLS_PER_CHUNK columns in 16-wide WMMA tiles, K=512.
__global__ __launch_bounds__(128) void wmma_lse_kernel(
    const unsigned short* __restrict__ ieb, const unsigned short* __restrict__ leb,
    const float* __restrict__ gps, float* __restrict__ part) {
  const int lane = threadIdx.x & 31;
  const int wave = threadIdx.x >> 5;
  const int hi = lane >> 4;     // which half-wave
  const int m16 = lane & 15;
  const int rowbase = blockIdx.x * 64 + wave * 16;
  const int chunk = blockIdx.y;
  const int col0 = chunk * COLS_PER_CHUNK;

  // Preload all 16 A fragments (16 rows x 512 K) -> registers, reused 128x.
  // ISA A layout (16-bit 16x32): lane m16 holds row rowbase+m16,
  // K = k0 + 8*hi + [0..7]  and  K = k0 + 16 + 8*hi + [0..7].
  Frag16 a[16];
  {
    const unsigned short* arow = ieb + (size_t)(rowbase + m16) * DD + hi * 8;
#pragma unroll
    for (int k = 0; k < 16; ++k) {
      a[k].q[0] = *(const uint4*)(arow + k * 32);
      a[k].q[1] = *(const uint4*)(arow + k * 32 + 16);
    }
  }
  // gps of this lane's 8 C-rows (C layout: M = v + 8*hi)
  float gxr[8], gyr[8];
#pragma unroll
  for (int v = 0; v < 8; ++v) {
    float2 g = *(const float2*)(gps + 2 * (rowbase + v + 8 * hi));
    gxr[v] = g.x; gyr[v] = g.y;
  }

  float s[8];
#pragma unroll
  for (int v = 0; v < 8; ++v) s[v] = 0.0f;

  for (int t = 0; t < COLS_PER_CHUNK / 16; ++t) {
    const int col = col0 + t * 16 + m16;  // this lane's column (B and C agree: N = lane&15)
    // B layout (32x16 bf16): lane holds column `col`, K = k0 + 8*hi + [0..7] and k0 + 16 + 8*hi + [0..7]
    const unsigned short* brow = leb + (size_t)col * DD + hi * 8;
    v8f c = {};
#pragma unroll
    for (int k = 0; k < 16; ++k) {
      Frag16 b;
      b.q[0] = *(const uint4*)(brow + k * 32);
      b.q[1] = *(const uint4*)(brow + k * 32 + 16);
      c = __builtin_amdgcn_wmma_f32_16x16x32_bf16(
          false, a[k].v, false, b.v, (short)0, c, false, false);
    }
    const float2 gj = *(const float2*)(gps + 2 * col);
#pragma unroll
    for (int v = 0; v < 8; ++v) {
      const float dx = gxr[v] - gj.x;
      const float dy = gyr[v] - gj.y;
      const bool far = fmaf(dx, dx, dy * dy) > 1e-8f;   // sqrt(d2) > 1e-4
      const float e = __builtin_amdgcn_exp2f(fmaf(c[v], KSC, -KSC));  // exp(sim/T - M0)
      s[v] += far ? e : 0.0f;
    }
  }

  // Reduce each row's 16 column-partials across the half-wave.
#pragma unroll
  for (int v = 0; v < 8; ++v) {
    float x = s[v];
#pragma unroll
    for (int off = 1; off < 16; off <<= 1) x += __shfl_xor(x, off, 32);
    s[v] = x;
  }
  if (m16 == 0) {  // lanes 0 and 16 each write 8 rows
    float* dst = part + (size_t)chunk * BB + rowbase + 8 * hi;
#pragma unroll
    for (int v = 0; v < 8; ++v) dst[v] = s[v];
  }
}

// ---------------- kernel 3: deterministic final reduction ----------------
__global__ __launch_bounds__(1024) void combine_kernel(
    const float* __restrict__ part, const float* __restrict__ pos,
    float* __restrict__ out) {
  __shared__ float lsum[1024];
  __shared__ float lcnt[1024];
  const int t = threadIdx.x;
  float acc = 0.f, cnt = 0.f;
  for (int r = t; r < BB; r += 1024) {
    float s = 0.f;
#pragma unroll
    for (int c = 0; c < NCHUNK; ++c) s += part[(size_t)c * BB + r];
    if (s > 0.f) {                               // row has at least one far negative
      const float p = pos[r];
      const float e0 = exp2f((p - M0F) * L2E);   // exp(pos - M0)
      const float lse = M0F + log2f(e0 + s) * LN2F;
      acc += lse - p;
      cnt += 1.f;
    }
  }
  lsum[t] = acc; lcnt[t] = cnt;
  __syncthreads();
  for (int off = 512; off > 0; off >>= 1) {
    if (t < off) { lsum[t] += lsum[t + off]; lcnt[t] += lcnt[t + off]; }
    __syncthreads();
  }
  if (t == 0) out[0] = lsum[0] / fmaxf(lcnt[0], 1.0f);
}

// ---------------- host launch ----------------
extern "C" void kernel_launch(void* const* d_in, const int* in_sizes, int n_in,
                              void* d_out, int out_size, void* d_ws, size_t ws_size,
                              hipStream_t stream) {
  const float* ie = (const float*)d_in[0];
  const float* le = (const float*)d_in[1];
  const float* gps = (const float*)d_in[2];

  char* ws = (char*)d_ws;
  unsigned short* ieb = (unsigned short*)ws;                         // 8 MB
  unsigned short* leb = (unsigned short*)(ws + (size_t)BB * DD * 2); // 8 MB
  float* pos = (float*)(ws + (size_t)BB * DD * 4);                   // 32 KB
  float* part = (float*)(ws + (size_t)BB * DD * 4 + (size_t)BB * 4); // 128 KB

  norm_pos_kernel<<<BB / 8, 256, 0, stream>>>(ie, le, ieb, leb, pos);
  wmma_lse_kernel<<<dim3(BB / 64, NCHUNK), 128, 0, stream>>>(ieb, leb, gps, part);
  combine_kernel<<<1, 1024, 0, stream>>>(part, pos, (float*)d_out);
}
